// AttnNetwork_51213190037935
// MI455X (gfx1250) — compile-verified, matrix operand order checked by reading, not run on hardware
//
#include <hip/hip_runtime.h>
#include <hip/hip_bf16.h>
#include <math.h>

// ---------------------------------------------------------------------------
// GNMT-style attention net on MI455X (gfx1250, wave32, WMMA bf16).
// All GEMMs: D = A[M,K] * B[N,K]^T (+bias/+Cin, opt tanh) via
// v_wmma_f32_16x16x32_bf16. fp32->bf16 conversion happens during LDS staging
// with float4 global loads + packed bf16 stores (ds_store_b64).
// Vocab projection fused with online-softmax NLL (no 262MB pred buffer).
// ---------------------------------------------------------------------------

typedef __bf16 bf16_t;
typedef __attribute__((ext_vector_type(16))) __bf16 v16bf;
typedef __attribute__((ext_vector_type(8)))  float  v8f;

#define HD   512
#define BB   64
#define SEQ  32
#define TDD  32     // decoder steps (T-1)
#define VOC  32000

#define WMMA_BF16(a,b,c) \
  __builtin_amdgcn_wmma_f32_16x16x32_bf16(false,(a),false,(b),(short)0,(c),false,false)

__device__ __forceinline__ bf16_t f2bf(float x){
  union { float f; unsigned u; } v; v.f = x;
  unsigned r = v.u + 0x7FFFu + ((v.u >> 16) & 1u);   // round-to-nearest-even
  union { unsigned short s; bf16_t b; } o; o.s = (unsigned short)(r >> 16);
  return o.b;
}

// pack two fp32 -> two bf16 (RNE) in one dword
__device__ __forceinline__ unsigned pack2bf(float a, float b){
  union { float f; unsigned u; } ua, ub; ua.f = a; ub.f = b;
  unsigned ra = ua.u + 0x7FFFu + ((ua.u >> 16) & 1u);
  unsigned rb = ub.u + 0x7FFFu + ((ub.u >> 16) & 1u);
  return (ra >> 16) | (rb & 0xFFFF0000u);
}
__device__ __forceinline__ uint2 pack4bf(float4 v){
  return make_uint2(pack2bf(v.x, v.y), pack2bf(v.z, v.w));
}

// Load a 16x32 bf16 operand fragment from row-major LDS tile.
// ISA 7.12.2 16-bit operand layout: lanes 0-15 row=lane, K{0..7,16..23};
// lanes 16-31 row=lane&15, K{8..15,24..31}.  Lowers to 2x ds_load_b128.
__device__ __forceinline__ v16bf ldfrag(const bf16_t* base, int ldst, int row0, int lane){
  int r  = row0 + (lane & 15);
  int kb = (lane >> 4) * 8;
  const bf16_t* p = base + r * ldst + kb;
  v16bf f;
#pragma unroll
  for (int j = 0; j < 8; ++j) f[j]     = p[j];
#pragma unroll
  for (int j = 0; j < 8; ++j) f[8 + j] = p[16 + j];
  return f;
}

// ---------------------------------------------------------------------------
// Generic GEMM: D[M,N] = A[M,K] * B[N,K]^T (+bias[N]) (+Cin[M,N]) (opt tanh)
// 64x64 tile per block, 128 threads (4 waves), wave w -> rows [16w,16w+16).
// M,N multiples of 64; K multiple of 32 (true for all call sites).
// ---------------------------------------------------------------------------
__global__ __launch_bounds__(128)
void gemm_tn_kernel(const float* __restrict__ A, int lda,
                    const float* __restrict__ Bm, int ldb,
                    float* __restrict__ D, int ldd,
                    const float* __restrict__ bias,
                    const float* __restrict__ Cin, int ldc,
                    int K, int act){
  __shared__ bf16_t As[64][48];
  __shared__ bf16_t Bs[64][48];
  const int tid  = threadIdx.x;
  const int lane = tid & 31;
  const int w    = tid >> 5;
  const int n0   = blockIdx.x * 64;
  const int m0   = blockIdx.y * 64;
  v8f acc[4] = {};
  for (int k0 = 0; k0 < K; k0 += 32){
    // stage 64x32 fp32 -> bf16 : 512 float4 loads over 128 threads
#pragma unroll
    for (int i = 0; i < 4; ++i){
      int idx = i * 128 + tid;            // 0..511
      int r   = idx >> 3;                 // 8 float4 per row
      int c4  = (idx & 7) * 4;
      float4 va = *(const float4*)&A [(size_t)(m0 + r) * lda + k0 + c4];
      float4 vb = *(const float4*)&Bm[(size_t)(n0 + r) * ldb + k0 + c4];
      *(uint2*)&As[r][c4] = pack4bf(va);  // ds_store_b64
      *(uint2*)&Bs[r][c4] = pack4bf(vb);
    }
    __syncthreads();
    if (k0 + 32 < K){   // hint next chunk toward L2 (global_prefetch_b8)
      __builtin_prefetch(&A [(size_t)(m0 + (tid >> 1)) * lda + k0 + 32], 0, 1);
      __builtin_prefetch(&Bm[(size_t)(n0 + (tid >> 1)) * ldb + k0 + 32], 0, 1);
    }
    v16bf af = ldfrag(&As[0][0], 48, w * 16, lane);
#pragma unroll
    for (int nt = 0; nt < 4; ++nt){
      v16bf bf = ldfrag(&Bs[0][0], 48, nt * 16, lane);
      acc[nt] = WMMA_BF16(af, bf, acc[nt]);
    }
    __syncthreads();
  }
  // C/D layout: lane<16 -> N=lane, M=i ; lane>=16 -> N=lane-16, M=8+i
#pragma unroll
  for (int nt = 0; nt < 4; ++nt){
    int n     = n0 + nt * 16 + (lane & 15);
    int mbase = m0 + w * 16 + ((lane < 16) ? 0 : 8);
#pragma unroll
    for (int i = 0; i < 8; ++i){
      int m = mbase + i;
      float v = acc[nt][i];
      if (bias) v += bias[n];
      if (Cin)  v += Cin[(size_t)m * ldc + n];
      if (act == 1) v = tanhf(v);
      D[(size_t)m * ldd + n] = v;
    }
  }
}

// ---------------------------------------------------------------------------
// LSTM pointwise gates (PyTorch i,f,g,o order).
// ---------------------------------------------------------------------------
__global__ void lstm_gates_kernel(const float* __restrict__ g,
                                  float* __restrict__ c_state,
                                  float* __restrict__ h_state,
                                  float* __restrict__ hs_out, int ld_out){
  int idx = blockIdx.x * blockDim.x + threadIdx.x;  // [0, B*H)
  int b = idx >> 9;
  int h = idx & 511;
  const float* gr = g + (size_t)b * 4 * HD;
  float i_ = gr[h], f_ = gr[HD + h], gg = gr[2 * HD + h], o_ = gr[3 * HD + h];
  float si = 1.f / (1.f + expf(-i_));
  float sf = 1.f / (1.f + expf(-f_));
  float so = 1.f / (1.f + expf(-o_));
  float c  = sf * c_state[idx] + si * tanhf(gg);
  float hn = so * tanhf(c);
  c_state[idx] = c;
  h_state[idx] = hn;
  hs_out[(size_t)b * ld_out + h] = hn;
}

// out[(s*B+b)*512 + d] = emb[ids[b*ldi + s]*512 + d]   (time-major activations)
__global__ void gather_embed_kernel(float* __restrict__ out, const float* __restrict__ emb,
                                    const int* __restrict__ ids, int ldi, int total){
  int e = blockIdx.x * blockDim.x + threadIdx.x;
  if (e >= total) return;
  int d  = e & 511;
  int rb = e >> 9;
  int b  = rb & 63;
  int s  = rb >> 6;
  int id = ids[b * ldi + s];
  out[e] = emb[(size_t)id * 512 + d];
}

__global__ void copy_cols_kernel(float* __restrict__ dst, int ldd, int off,
                                 const float* __restrict__ src, int lds_, int cols, int total){
  int e = blockIdx.x * blockDim.x + threadIdx.x;
  if (e >= total) return;
  int r = e / cols, c = e % cols;
  dst[(size_t)r * ldd + off + c] = src[(size_t)r * lds_ + c];
}

__global__ void zero_kernel(float* p, int n){
  int i = blockIdx.x * blockDim.x + threadIdx.x;
  if (i < n) p[i] = 0.f;
}

// ---------------------------------------------------------------------------
// scores[b,s,t] = sum_k e_proj[b,s,k]*d_proj[b,t,k]   (M=N=32, K=1024)
// one wave per batch; 2x2 WMMA tiles.
// ---------------------------------------------------------------------------
__global__ __launch_bounds__(32)
void attn_scores_kernel(const float* __restrict__ eproj, const float* __restrict__ dproj,
                        float* __restrict__ scores){
  __shared__ bf16_t As[32][48];
  __shared__ bf16_t Bs[32][48];
  int b = blockIdx.x;
  int lane = threadIdx.x;
  v8f a00 = {}, a01 = {}, a10 = {}, a11 = {};
  for (int k0 = 0; k0 < 1024; k0 += 32){
    // 32x32 tile = 256 float4 over 32 lanes -> 8 each
#pragma unroll
    for (int i = 0; i < 8; ++i){
      int idx = i * 32 + lane;
      int r   = idx >> 3;
      int c4  = (idx & 7) * 4;
      float4 va = *(const float4*)&eproj[((size_t)r * BB + b) * 1024 + k0 + c4];
      float4 vb = *(const float4*)&dproj[((size_t)r * BB + b) * 1024 + k0 + c4];
      *(uint2*)&As[r][c4] = pack4bf(va);
      *(uint2*)&Bs[r][c4] = pack4bf(vb);
    }
    __syncthreads();
    v16bf fa0 = ldfrag(&As[0][0], 48, 0, lane), fa1 = ldfrag(&As[0][0], 48, 16, lane);
    v16bf fb0 = ldfrag(&Bs[0][0], 48, 0, lane), fb1 = ldfrag(&Bs[0][0], 48, 16, lane);
    a00 = WMMA_BF16(fa0, fb0, a00);  a01 = WMMA_BF16(fa0, fb1, a01);
    a10 = WMMA_BF16(fa1, fb0, a10);  a11 = WMMA_BF16(fa1, fb1, a11);
    __syncthreads();
  }
  v8f accs[2][2] = { { a00, a01 }, { a10, a11 } };
#pragma unroll
  for (int mt = 0; mt < 2; ++mt)
#pragma unroll
    for (int nt = 0; nt < 2; ++nt){
      int n  = nt * 16 + (lane & 15);
      int mb = mt * 16 + ((lane < 16) ? 0 : 8);
#pragma unroll
      for (int i = 0; i < 8; ++i)
        scores[(size_t)b * SEQ * TDD + (mb + i) * TDD + n] = accs[mt][nt][i];
    }
}

// softmax over s (axis=1): one wave per (b,t) column
__global__ __launch_bounds__(32)
void softmax_s_kernel(float* scores){
  int idx = blockIdx.x;
  int b = idx >> 5, t = idx & 31;
  int s = threadIdx.x;
  float* base = scores + (size_t)b * SEQ * TDD;
  float v = base[s * TDD + t];
  float m = v;
#pragma unroll
  for (int o = 16; o; o >>= 1) m = fmaxf(m, __shfl_xor(m, o, 32));
  float e = expf(v - m);
  float sum = e;
#pragma unroll
  for (int o = 16; o; o >>= 1) sum += __shfl_xor(sum, o, 32);
  base[s * TDD + t] = e / sum;
}

// context[t,k] = sum_s attn[s,t] * enc_h[s,k]; writes into cat cols [0,1024)
// grid (16 n-chunks, B batches), 64 threads (2 waves), K=S=32 -> one WMMA step
__global__ __launch_bounds__(64)
void attn_context_kernel(const float* __restrict__ scores, const float* __restrict__ ench,
                         float* __restrict__ cat, int ldcat){
  __shared__ bf16_t At[32][48];
  __shared__ bf16_t Bsh[32][72];
  int b  = blockIdx.y;
  int n0 = blockIdx.x * 64;
  int tid = threadIdx.x, lane = tid & 31, w = tid >> 5;
  // A = attn^T : load float4 along t, scatter into At[t..t+3][s]
#pragma unroll
  for (int i = 0; i < 4; ++i){
    int idx = i * 64 + tid;                 // 0..255 float4s
    int s   = idx >> 3;
    int t4  = (idx & 7) * 4;
    float4 v = *(const float4*)&scores[(size_t)b * SEQ * TDD + s * TDD + t4];
    At[t4 + 0][s] = f2bf(v.x);
    At[t4 + 1][s] = f2bf(v.y);
    At[t4 + 2][s] = f2bf(v.z);
    At[t4 + 3][s] = f2bf(v.w);
  }
  // B (K=s, N) tile: 32x64 = 512 float4 over 64 threads -> 8 each
#pragma unroll
  for (int i = 0; i < 8; ++i){
    int idx = i * 64 + tid;
    int s   = idx >> 4;                     // 16 float4 per row
    int n4  = (idx & 15) * 4;
    float4 v = *(const float4*)&ench[((size_t)s * BB + b) * 1024 + n0 + n4];
    *(uint2*)&Bsh[s][n4] = pack4bf(v);
  }
  __syncthreads();
  v16bf a = ldfrag(&At[0][0], 48, w * 16, lane);
  v8f acc[4] = {};
#pragma unroll
  for (int nt = 0; nt < 4; ++nt){
    int col = nt * 16 + (lane & 15);
    int kb  = (lane >> 4) * 8;
    v16bf bf;
#pragma unroll
    for (int j = 0; j < 8; ++j) bf[j]     = Bsh[kb + j][col];
#pragma unroll
    for (int j = 0; j < 8; ++j) bf[8 + j] = Bsh[16 + kb + j][col];
    acc[nt] = WMMA_BF16(a, bf, acc[nt]);
  }
#pragma unroll
  for (int nt = 0; nt < 4; ++nt){
    int n     = n0 + nt * 16 + (lane & 15);
    int tbase = w * 16 + ((lane < 16) ? 0 : 8);
#pragma unroll
    for (int i = 0; i < 8; ++i)
      cat[((size_t)(tbase + i) * BB + b) * ldcat + n] = acc[nt][i];
  }
}

// ---------------------------------------------------------------------------
// Fused vocab projection + online-softmax NLL.
// Block owns 64 rows of t_tm (staged bf16 in dynamic LDS, 128KB of 320KB WGP
// LDS), streams W_C in 64-col chunks via WMMA (B fragments loaded directly
// from global as float4 -> bf16). Running (max,sumexp,label logit) per row.
// ---------------------------------------------------------------------------
__global__ __launch_bounds__(128)
void proj_loss_kernel(const float* __restrict__ Tm, const float* __restrict__ WC,
                      const float* __restrict__ bC, const int* __restrict__ y,
                      float* __restrict__ nll_out){
  extern __shared__ unsigned char smem[];
  bf16_t* Asm  = (bf16_t*)smem;                                       // 64 x 1032
  float* chunk = (float*)(smem + (size_t)64 * 1032 * sizeof(bf16_t)); // 64 x 65
  float* mrun  = chunk + 64 * 65;
  float* srun  = mrun + 64;
  float* llog  = srun + 64;
  int*   lab   = (int*)(llog + 64);

  const int tid = threadIdx.x, lane = tid & 31, w = tid >> 5;
  const int m0  = blockIdx.x * 64;

  // stage 64x1024 fp32 rows -> bf16 : 16384 float4 over 128 threads
#pragma unroll 4
  for (int i = 0; i < 128; ++i){
    int idx = i * 128 + tid;
    int r   = idx >> 8;                    // 256 float4 per row
    int c4  = (idx & 255) * 4;
    float4 v = *(const float4*)&Tm[(size_t)(m0 + r) * 1024 + c4];
    *(uint2*)&Asm[r * 1032 + c4] = pack4bf(v);
  }
  if (tid < 64){
    int row = m0 + tid;                    // row = t*B + b
    int b = row & 63, t = row >> 6;
    lab[tid]  = y[b * 33 + t + 1];         // labels = y[:,1:]
    mrun[tid] = -INFINITY; srun[tid] = 0.f; llog[tid] = 0.f;
  }
  __syncthreads();

  for (int n0 = 0; n0 < VOC; n0 += 64){
    v8f acc[4] = {};
    for (int ks = 0; ks < 32; ++ks){
      v16bf af = ldfrag(Asm + ks * 32, 1032, w * 16, lane);
#pragma unroll
      for (int nt = 0; nt < 4; ++nt){
        int n  = n0 + nt * 16 + (lane & 15);
        int kb = ks * 32 + (lane >> 4) * 8;
        const float4* p0 = (const float4*)(WC + (size_t)n * 1024 + kb);
        const float4* p1 = (const float4*)(WC + (size_t)n * 1024 + kb + 16);
        float4 x0 = p0[0], x1 = p0[1], y0 = p1[0], y1 = p1[1];
        v16bf bf;
        bf[0]=f2bf(x0.x); bf[1]=f2bf(x0.y); bf[2]=f2bf(x0.z); bf[3]=f2bf(x0.w);
        bf[4]=f2bf(x1.x); bf[5]=f2bf(x1.y); bf[6]=f2bf(x1.z); bf[7]=f2bf(x1.w);
        bf[8]=f2bf(y0.x); bf[9]=f2bf(y0.y); bf[10]=f2bf(y0.z); bf[11]=f2bf(y0.w);
        bf[12]=f2bf(y1.x); bf[13]=f2bf(y1.y); bf[14]=f2bf(y1.z); bf[15]=f2bf(y1.w);
        acc[nt] = WMMA_BF16(af, bf, acc[nt]);
      }
    }
#pragma unroll
    for (int nt = 0; nt < 4; ++nt){
      int nloc = nt * 16 + (lane & 15);
      float bias = bC[n0 + nloc];
      int rbase = w * 16 + ((lane < 16) ? 0 : 8);
#pragma unroll
      for (int i = 0; i < 8; ++i)
        chunk[(rbase + i) * 65 + nloc] = acc[nt][i] + bias;
    }
    __syncthreads();
    if (tid < 64){
      int r = tid;
      float cm = -INFINITY;
#pragma unroll 1
      for (int j = 0; j < 64; ++j) cm = fmaxf(cm, chunk[r * 65 + j]);
      float cs = 0.f;
#pragma unroll 1
      for (int j = 0; j < 64; ++j) cs += expf(chunk[r * 65 + j] - cm);
      float m_old = mrun[r];
      float m_new = fmaxf(m_old, cm);
      srun[r] = srun[r] * expf(m_old - m_new) + cs * expf(cm - m_new);
      mrun[r] = m_new;
      int lb = lab[r];
      if (lb >= n0 && lb < n0 + 64) llog[r] = chunk[r * 65 + (lb - n0)];
    }
    __syncthreads();
  }
  if (tid < 64)
    nll_out[m0 + tid] = (mrun[tid] + logf(srun[tid])) - llog[tid];
}

// deterministic final reduction: loss = sum(nll)/B
__global__ __launch_bounds__(256)
void reduce_loss_kernel(const float* __restrict__ nll, float* __restrict__ out){
  __shared__ float sh[256];
  int tid = threadIdx.x;
  float s = 0.f;
  for (int i = tid; i < TDD * BB; i += 256) s += nll[i];
  sh[tid] = s; __syncthreads();
  for (int o = 128; o; o >>= 1){ if (tid < o) sh[tid] += sh[tid + o]; __syncthreads(); }
  if (tid == 0) out[0] = sh[0] * (1.0f / (float)BB);
}

// ---------------------------------------------------------------------------
extern "C" void kernel_launch(void* const* d_in, const int* in_sizes, int n_in,
                              void* d_out, int out_size, void* d_ws, size_t ws_size,
                              hipStream_t stream) {
  (void)in_sizes; (void)n_in; (void)out_size; (void)ws_size;
  const int*   x      = (const int*)d_in[0];
  const int*   y      = (const int*)d_in[1];
  const float* emb_de = (const float*)d_in[2];
  const float* emb_en = (const float*)d_in[3];
  const float* encW[2][2][4];   // [layer][fwd/bwd][Wih,Whh,bih,bhh]
  int k = 4;
  for (int l = 0; l < 2; ++l)
    for (int d = 0; d < 2; ++d)
      for (int p = 0; p < 4; ++p) encW[l][d][p] = (const float*)d_in[k++];
  const float* decW[4][4];
  for (int l = 0; l < 4; ++l)
    for (int p = 0; p < 4; ++p) decW[l][p] = (const float*)d_in[k++];
  const float* W_E = (const float*)d_in[36];
  const float* b_E = (const float*)d_in[37];
  const float* W_D = (const float*)d_in[38];
  const float* b_D = (const float*)d_in[39];
  const float* W_A = (const float*)d_in[40];
  const float* b_A = (const float*)d_in[41];
  const float* W_C = (const float*)d_in[42];
  const float* b_C = (const float*)d_in[43];

  // ---- workspace carve (time-major activations [t][b][feat]) ----
  char* p = (char*)d_ws;
  auto bump = [&](size_t n){ float* r = (float*)p; p += ((n * sizeof(float) + 255) / 256) * 256; return r; };
  float* ex    = bump((size_t)SEQ * BB * 512);
  float* ey    = bump((size_t)TDD * BB * 512);
  float* cat0  = bump((size_t)SEQ * BB * 1024);   // enc layer0 fwd|bwd concat
  float* cat1  = bump((size_t)SEQ * BB * 1024);   // enc_h
  float* xw    = bump((size_t)SEQ * BB * 2048);   // input-transform buffer
  float* g     = bump((size_t)BB * 2048);         // per-step gate pre-act
  float* st_h[4]; float* st_c[4];                 // enc finals == dec inits
  for (int i = 0; i < 4; ++i){ st_h[i] = bump(BB * HD); st_c[i] = bump(BB * HD); }
  float* dhs0  = bump((size_t)TDD * BB * 512);
  float* dhs1  = bump((size_t)TDD * BB * 512);
  float* eproj = bump((size_t)SEQ * BB * 1024);
  float* dproj = bump((size_t)TDD * BB * 1024);
  float* scor  = bump((size_t)BB * SEQ * TDD);
  float* catA  = bump((size_t)TDD * BB * 1536);   // [context | dec_h]
  float* ttm   = bump((size_t)TDD * BB * 1024);
  float* nll   = bump((size_t)TDD * BB);

  auto gemm = [&](float* D, int ldd, const float* A, int lda, const float* Bm, int ldb,
                  int M, int N, int K, const float* bias, const float* Cin, int ldc, int act){
    dim3 grid(N / 64, M / 64);
    gemm_tn_kernel<<<grid, 128, 0, stream>>>(A, lda, Bm, ldb, D, ldd, bias, Cin, ldc, K, act);
  };

  // ---- embeddings (time-major) ----
  gather_embed_kernel<<<(SEQ * BB * 512) / 256, 256, 0, stream>>>(ex, emb_de, x, 32, SEQ * BB * 512);
  gather_embed_kernel<<<(TDD * BB * 512) / 256, 256, 0, stream>>>(ey, emb_en, y, 33, TDD * BB * 512);

  for (int i = 0; i < 4; ++i){
    zero_kernel<<<(BB * HD + 255) / 256, 256, 0, stream>>>(st_h[i], BB * HD);
    zero_kernel<<<(BB * HD + 255) / 256, 256, 0, stream>>>(st_c[i], BB * HD);
  }

  // ---- encoder: 2 bidirectional layers ----
  const float* enc_in[2]  = { ex, cat0 };
  int          enc_k[2]   = { 512, 1024 };
  float*       enc_out[2] = { cat0, cat1 };
  for (int l = 0; l < 2; ++l){
    for (int d = 0; d < 2; ++d){
      int sidx = l * 2 + d;                       // matches h_fin order l0f,l0b,l1f,l1b
      const float* Wih = encW[l][d][0]; const float* Whh = encW[l][d][1];
      const float* bih = encW[l][d][2]; const float* bhh = encW[l][d][3];
      gemm(xw, 2048, enc_in[l], enc_k[l], Wih, enc_k[l], SEQ * BB, 2048, enc_k[l], bih, nullptr, 0, 0);
      for (int s = 0; s < SEQ; ++s){
        int t = d ? (SEQ - 1 - s) : s;
        gemm(g, 2048, st_h[sidx], 512, Whh, 512, BB, 2048, 512,
             bhh, xw + (size_t)t * BB * 2048, 2048, 0);
        lstm_gates_kernel<<<(BB * HD) / 256, 256, 0, stream>>>(
            g, st_c[sidx], st_h[sidx],
            enc_out[l] + (size_t)t * BB * 1024 + d * 512, 1024);
      }
    }
  }

  // ---- decoder: 4 layers, init states = encoder finals (in place) ----
  const float* din = ey;
  float* douts[4] = { dhs0, dhs1, dhs0, dhs1 };
  for (int l = 0; l < 4; ++l){
    const float* Wih = decW[l][0]; const float* Whh = decW[l][1];
    const float* bih = decW[l][2]; const float* bhh = decW[l][3];
    gemm(xw, 2048, din, 512, Wih, 512, TDD * BB, 2048, 512, bih, nullptr, 0, 0);
    for (int t = 0; t < TDD; ++t){
      gemm(g, 2048, st_h[l], 512, Whh, 512, BB, 2048, 512,
           bhh, xw + (size_t)t * BB * 2048, 2048, 0);
      lstm_gates_kernel<<<(BB * HD) / 256, 256, 0, stream>>>(
          g, st_c[l], st_h[l], douts[l] + (size_t)t * BB * 512, 512);
    }
    din = douts[l];
  }
  float* dec_h = dhs1;

  // ---- attention ----
  gemm(eproj, 1024, cat1, 1024, W_E, 1024, SEQ * BB, 1024, 1024, b_E, nullptr, 0, 0);
  gemm(dproj, 1024, dec_h, 512, W_D, 512, TDD * BB, 1024, 512, b_D, nullptr, 0, 0);
  attn_scores_kernel<<<BB, 32, 0, stream>>>(eproj, dproj, scor);
  softmax_s_kernel<<<BB * TDD, 32, 0, stream>>>(scor);
  attn_context_kernel<<<dim3(16, BB), 64, 0, stream>>>(scor, cat1, catA, 1536);
  copy_cols_kernel<<<(TDD * BB * 512) / 256, 256, 0, stream>>>(
      catA, 1536, 1024, dec_h, 512, 512, TDD * BB * 512);
  gemm(ttm, 1024, catA, 1536, W_A, 1536, TDD * BB, 1024, 1536, b_A, nullptr, 0, 1);

  // ---- fused vocab projection + NLL ----
  size_t smem = (size_t)64 * 1032 * sizeof(bf16_t) + 64 * 65 * 4 + 3 * 64 * 4 + 64 * 4;
  proj_loss_kernel<<<(TDD * BB) / 64, 128, smem, stream>>>(ttm, W_C, b_C, y, nll);
  reduce_loss_kernel<<<1, 256, 0, stream>>>(nll, (float*)d_out);
}
